// DotAttn_20083267076209
// MI455X (gfx1250) — matrix operand order repaired and checked by reading, not compiled, not run on hardware
//
#include <hip/hip_runtime.h>
#include <hip/hip_fp16.h>

typedef __attribute__((ext_vector_type(16))) _Float16 v16h;
typedef __attribute__((ext_vector_type(8)))  _Float16 v8h;
typedef __attribute__((ext_vector_type(4)))  _Float16 v4h;
typedef __attribute__((ext_vector_type(8)))  float    v8f;
typedef __attribute__((ext_vector_type(4)))  float    v4f;

#define QLEN 512
#define NS   64
#define LS   128
#define ED   256
#define QTILE 128
#define KSTRIDE 264   // halves per K LDS row (256 + 8 pad)
#define VTSTRIDE 136  // halves per V^T LDS row (128 + 8 pad)
#define PSTRIDE 136   // halves per P LDS row

#define K_OFF   0
#define VT_OFF  (LS * KSTRIDE * 2)                 // 67584
#define PEN_OFF (VT_OFF + ED * VTSTRIDE * 2)       // 67584 + 69632
#define SMEM_BYTES (PEN_OFF + LS * 4)              // 137728 bytes dynamic LDS

__global__ void __launch_bounds__(256)
attn_snippet_kernel(const float* __restrict__ q, const float* __restrict__ k,
                    const float* __restrict__ v, const unsigned char* __restrict__ mask,
                    float* __restrict__ out)
{
    extern __shared__ char smem[];
    _Float16* kl  = (_Float16*)(smem + K_OFF);    // K tile f16 [128][KSTRIDE]
    _Float16* pl  = (_Float16*)(smem + K_OFF);    // P tile f16 [128][PSTRIDE] (aliases K)
    _Float16* vt  = (_Float16*)(smem + VT_OFF);   // V^T f16 [256][VTSTRIDE]
    float*    pen = (float*)(smem + PEN_OFF);     // penalty per key

    const int t    = threadIdx.x;
    const int lane = t & 31;
    const int wave = t >> 5;
    const int mrow = lane & 15;   // M (or N) index within 16
    const int hi   = lane >> 4;   // half-wave select
    const int qt   = blockIdx.x;  // 0..3  query tile
    const int s    = blockIdx.y;  // 0..63 snippet
    const int b    = blockIdx.z;  // 0..7  batch

    const float* kg = k + (size_t)(b * NS + s) * LS * ED;
    const float* vg = v + (size_t)(b * NS + s) * LS * ED;
    const unsigned char* mg = mask + (size_t)(b * NS + s) * LS;

    // ---- Stage K (f16 row-major) and V^T (f16) into LDS with f32->f16 convert.
    // 128*256 f32 = 8192 float4 per tensor; 256 threads * 32 iters, coalesced.
    for (int i = 0; i < 32; ++i) {
        int idx4 = i * 256 + t;
        int l  = idx4 >> 6;       // row (key index)
        int e4 = idx4 & 63;       // float4 index along e
        v4f kv = *(const v4f*)(kg + (size_t)l * ED + e4 * 4);
        v4h kh; kh.x = (_Float16)kv.x; kh.y = (_Float16)kv.y;
                kh.z = (_Float16)kv.z; kh.w = (_Float16)kv.w;
        *(v4h*)(kl + (size_t)l * KSTRIDE + e4 * 4) = kh;

        v4f vv = *(const v4f*)(vg + (size_t)l * ED + e4 * 4);
        int e = e4 * 4;
        vt[(size_t)(e + 0) * VTSTRIDE + l] = (_Float16)vv.x;
        vt[(size_t)(e + 1) * VTSTRIDE + l] = (_Float16)vv.y;
        vt[(size_t)(e + 2) * VTSTRIDE + l] = (_Float16)vv.z;
        vt[(size_t)(e + 3) * VTSTRIDE + l] = (_Float16)vv.w;
    }
    if (t < LS) pen[t] = mg[t] ? 0.0f : -1.0e12f;
    __syncthreads();

    // ---- Phase 1: S = Q * K^T for this wave's 16 query rows (16x128 in 8 tiles).
    const float* qg = q + (size_t)(b * QLEN + qt * QTILE + wave * 16 + mrow) * ED;

    v8f acc[8] = {};
    for (int c = 0; c < 8; ++c) {               // e-chunks of 32
        int e0 = c * 32 + hi * 8;
        v4f q0 = *(const v4f*)(qg + e0);
        v4f q1 = *(const v4f*)(qg + e0 + 4);
        v4f q2 = *(const v4f*)(qg + e0 + 16);
        v4f q3 = *(const v4f*)(qg + e0 + 20);
        v16h a;
#pragma unroll
        for (int j = 0; j < 4; ++j) {
            a[j]      = (_Float16)q0[j];
            a[4 + j]  = (_Float16)q1[j];
            a[8 + j]  = (_Float16)q2[j];
            a[12 + j] = (_Float16)q3[j];
        }
#pragma unroll
        for (int n = 0; n < 8; ++n) {           // key tiles of 16
            const _Float16* bp = kl + (size_t)(n * 16 + mrow) * KSTRIDE + c * 32 + hi * 8;
            v16h bf;
            *(v8h*)&bf       = *(const v8h*)bp;
            *((v8h*)&bf + 1) = *(const v8h*)(bp + 16);
            acc[n] = __builtin_amdgcn_wmma_f32_16x16x32_f16(
                false, a, false, bf, (short)0, acc[n], false, false);
        }
    }

    // ---- Masked softmax over the 128 keys, row-wise (rows split across lane halves).
    float pn[8];
#pragma unroll
    for (int n = 0; n < 8; ++n) pn[n] = pen[n * 16 + mrow];

#pragma unroll
    for (int r = 0; r < 8; ++r) {
        float mx = -3.0e38f;
#pragma unroll
        for (int n = 0; n < 8; ++n) { acc[n][r] += pn[n]; mx = fmaxf(mx, acc[n][r]); }
#pragma unroll
        for (int d = 1; d < 16; d <<= 1) mx = fmaxf(mx, __shfl_xor(mx, d, 16));
        float sum = 0.0f;
#pragma unroll
        for (int n = 0; n < 8; ++n) { float e_ = __expf(acc[n][r] - mx); acc[n][r] = e_; sum += e_; }
#pragma unroll
        for (int d = 1; d < 16; d <<= 1) sum += __shfl_xor(sum, d, 16);
        float inv = 1.0f / sum;
#pragma unroll
        for (int n = 0; n < 8; ++n) acc[n][r] *= inv;
    }

    __syncthreads();   // all waves done reading K LDS; safe to overwrite with P

    // ---- Spill P (f16) to LDS; each wave re-reads only its own 16 rows.
#pragma unroll
    for (int r = 0; r < 8; ++r) {
        int m = wave * 16 + (hi ? r + 8 : r);
#pragma unroll
        for (int n = 0; n < 8; ++n)
            pl[(size_t)m * PSTRIDE + n * 16 + mrow] = (_Float16)acc[n][r];
    }
    asm volatile("s_wait_dscnt 0" ::: "memory");  // wave-local P stores -> loads

    // ---- Phase 2: O = P * V (16 x 256 per wave), two 128-wide halves.
    const size_t obase = ((size_t)(b * QLEN + qt * QTILE + wave * 16) * NS + s) * ED;
    for (int h = 0; h < 2; ++h) {
        v8f oacc[8] = {};
        for (int kk = 0; kk < 4; ++kk) {        // l-chunks of 32
            const _Float16* ap = pl + (size_t)(wave * 16 + mrow) * PSTRIDE + kk * 32 + hi * 8;
            v16h a;
            *(v8h*)&a       = *(const v8h*)ap;
            *((v8h*)&a + 1) = *(const v8h*)(ap + 16);
#pragma unroll
            for (int n = 0; n < 8; ++n) {       // e tiles of 16
                const _Float16* bp = vt + (size_t)(h * 128 + n * 16 + mrow) * VTSTRIDE
                                        + kk * 32 + hi * 8;
                v16h bf;
                *(v8h*)&bf       = *(const v8h*)bp;
                *((v8h*)&bf + 1) = *(const v8h*)(bp + 16);
                oacc[n] = __builtin_amdgcn_wmma_f32_16x16x32_f16(
                    false, a, false, bf, (short)0, oacc[n], false, false);
            }
        }
#pragma unroll
        for (int n = 0; n < 8; ++n) {
#pragma unroll
            for (int r = 0; r < 8; ++r) {
                int m = hi ? r + 8 : r;
                out[obase + (size_t)m * (NS * ED) + h * 128 + n * 16 + mrow] = oacc[n][r];
            }
        }
    }
}

extern "C" void kernel_launch(void* const* d_in, const int* in_sizes, int n_in,
                              void* d_out, int out_size, void* d_ws, size_t ws_size,
                              hipStream_t stream) {
    (void)in_sizes; (void)n_in; (void)out_size; (void)d_ws; (void)ws_size;
    const float* q = (const float*)d_in[0];
    const float* k = (const float*)d_in[1];
    const float* v = (const float*)d_in[2];
    const unsigned char* mask = (const unsigned char*)d_in[3];  // jax bool = 1 byte
    float* out = (float*)d_out;
    dim3 grid(QLEN / QTILE, NS, 8);   // (q-tiles, snippets, batch)
    attn_snippet_kernel<<<grid, 256, SMEM_BYTES, stream>>>(q, k, v, mask, out);
}
